// CrossAttention_85624468013221
// MI455X (gfx1250) — matrix operand order repaired and checked by reading, not compiled, not run on hardware
//
#include <hip/hip_runtime.h>
#include <stdint.h>

#define BATCH 8
#define CCH 128
#define HH 128
#define WW 128
#define HWPIX (HH * WW)          // 16384
#define HEADS 4
#define DC (CCH / HEADS)         // 32
#define EPS_LN 1e-6f
#define EPS_NORM 1e-12f
#define TPX 32                   // pixels per LN tile

typedef __bf16 bf16;
typedef __bf16 v8bf  __attribute__((ext_vector_type(8)));
typedef __bf16 v16bf __attribute__((ext_vector_type(16)));
typedef float  v8f   __attribute__((ext_vector_type(8)));

// ---------------------------------------------------------------------------
// Pack the four 128x128 f32 weight matrices to bf16 (q_pw, k_pw, v_pw, proj_w)
// ---------------------------------------------------------------------------
__global__ void pack_w(const float* __restrict__ a, const float* __restrict__ b,
                       const float* __restrict__ c, const float* __restrict__ d,
                       bf16* __restrict__ o) {
  int i = blockIdx.x * 256 + threadIdx.x;          // 0 .. 16383
  o[i]             = (bf16)a[i];
  o[16384 + i]     = (bf16)b[i];
  o[2 * 16384 + i] = (bf16)c[i];
  o[3 * 16384 + i] = (bf16)d[i];
}

__global__ void zero_f32(float* __restrict__ p, int n) {
  int i = blockIdx.x * 256 + threadIdx.x;
  if (i < n) p[i] = 0.f;
}

// ---------------------------------------------------------------------------
// Channel LayerNorm over C for state & dark; single global read via LDS tile
// of (128 ch x 32 px) for both tensors. Packs bf16: LN(state), LN(dark), dark.
// ---------------------------------------------------------------------------
__global__ void __launch_bounds__(256)
ln_pack(const float* __restrict__ state, const float* __restrict__ dark,
        const float* __restrict__ nqw, const float* __restrict__ nqb,
        const float* __restrict__ nkw, const float* __restrict__ nkb,
        bf16* __restrict__ xq, bf16* __restrict__ xk, bf16* __restrict__ xv) {
  __shared__ float sstate[CCH * TPX];     // 16 KB
  __shared__ float sdark[CCH * TPX];      // 16 KB
  __shared__ float sms[TPX], sis[TPX], smd[TPX], sid[TPX];
  __shared__ float spar[4 * CCH];         // 2 KB

  int tid = threadIdx.x;
  for (int i = tid; i < CCH; i += 256) {
    spar[i]           = nqw[i];
    spar[CCH + i]     = nqb[i];
    spar[2 * CCH + i] = nkw[i];
    spar[3 * CCH + i] = nkb[i];
  }

  long pblk = (long)blockIdx.x * TPX;     // HW % TPX == 0 -> one b per block
  int b  = (int)(pblk / HWPIX);
  int p0 = (int)(pblk % HWPIX);
  const float* sp = state + (long)b * CCH * HWPIX + p0;
  const float* dp = dark  + (long)b * CCH * HWPIX + p0;

  // one coalesced global read of both tiles (wave = one 128B channel row)
  for (int i = tid; i < CCH * TPX; i += 256) {
    int c = i >> 5, px = i & (TPX - 1);
    sstate[i] = sp[(long)c * HWPIX + px];
    sdark[i]  = dp[(long)c * HWPIX + px];
  }
  __syncthreads();

  // per-pixel stats: threads 0-31 -> state, 32-63 -> dark (bank-conflict-free)
  if (tid < 64) {
    int px = tid & (TPX - 1);
    const float* base = (tid < TPX) ? sstate : sdark;
    float ss = 0.f, ssq = 0.f;
    for (int c = 0; c < CCH; ++c) {
      float x = base[c * TPX + px];
      ss += x; ssq += x * x;
    }
    const float inv = 1.f / (float)CCH;
    float m = ss * inv;
    float v = ssq * inv - m * m;
    float r = rsqrtf(v + EPS_LN);
    if (tid < TPX) { sms[px] = m; sis[px] = r; }
    else           { smd[px] = m; sid[px] = r; }
  }
  __syncthreads();

  bf16* xqp = xq + (long)b * CCH * HWPIX + p0;
  bf16* xkp = xk + (long)b * CCH * HWPIX + p0;
  bf16* xvp = xv + (long)b * CCH * HWPIX + p0;
  for (int i = tid; i < CCH * TPX; i += 256) {
    int c = i >> 5, px = i & (TPX - 1);
    float x1 = sstate[i];
    float x2 = sdark[i];
    float q = (x1 - sms[px]) * sis[px] * spar[c]           + spar[CCH + c];
    float k = (x2 - smd[px]) * sid[px] * spar[2 * CCH + c] + spar[3 * CCH + c];
    long o = (long)c * HWPIX + px;
    xqp[o] = (bf16)q;
    xkp[o] = (bf16)k;
    xvp[o] = (bf16)x2;
  }
}

// ---------------------------------------------------------------------------
// 1x1 conv as GEMM: Y[b](128 x HW) = W(128x128) * X[b](128 x HW), bf16 WMMA,
// f32 accumulate. Block = 8 waves; wave w owns M-rows [16w,16w+16), block owns
// 32 pixels. X tile staged transposed in LDS via b128 loads. Output is bf16
// (BF16OUT) for q/k/v, or f32 + residual for the final projection.
// ---------------------------------------------------------------------------
#define XS_STRIDE 136   // 128 + 8 bf16 pad, keeps 16B alignment

template <bool BF16OUT>
__global__ void __launch_bounds__(256)
gemm_c128(const bf16* __restrict__ Wp,     // 128x128 bf16, row-major (O, Cin)
          const bf16* __restrict__ X,      // (B, 128, HW) bf16
          const float* __restrict__ resid, // (B, 128, HW) f32 (only !BF16OUT)
          void* __restrict__ Yv) {
  __shared__ bf16 xs[32 * XS_STRIDE];

  int tid  = threadIdx.x;
  int lane = tid & 31;
  int wave = tid >> 5;
  int b    = blockIdx.y;
  int n0   = blockIdx.x * 32;

  // cooperative transpose-load of X tile: 128(k) x 32(n) -> xs[n][k]
  {
    const bf16* xb = X + (long)b * CCH * HWPIX + n0;
    int n8 = (tid & 3) * 8;
    int kb = tid >> 2;                       // 0..63
    #pragma unroll
    for (int it = 0; it < 2; ++it) {
      int k = kb + it * 64;
      v8bf v = *(const v8bf*)(xb + (long)k * HWPIX + n8);
      #pragma unroll
      for (int j = 0; j < 8; ++j) xs[(n8 + j) * XS_STRIDE + k] = v[j];
    }
  }
  __syncthreads();

  int mbase = wave * 16;
  const bf16* wrow = Wp + (long)(mbase + (lane & 15)) * CCH;
  __builtin_prefetch(wrow, 0, 3);

  int aoff  = (lane >> 4) * 8;    // A: K half-select per ISA 16-bit A layout
  int bkoff = (lane >> 4) * 8;    // B: K half-select matching A layout
  int bn    = lane & 15;

  v8f acc0 = {}, acc1 = {};
  #pragma unroll
  for (int ks = 0; ks < 4; ++ks) {
    int kstep = ks * 32;
    v8bf alo = *(const v8bf*)(wrow + kstep + aoff);
    v8bf ahi = *(const v8bf*)(wrow + kstep + 16 + aoff);
    v16bf a;
    #pragma unroll
    for (int i = 0; i < 8; ++i) { a[i] = alo[i]; a[8 + i] = ahi[i]; }

    const bf16* bp0 = &xs[bn * XS_STRIDE + kstep + bkoff];
    const bf16* bp1 = &xs[(16 + bn) * XS_STRIDE + kstep + bkoff];
    v8bf b0lo = *(const v8bf*)bp0, b0hi = *(const v8bf*)(bp0 + 16);
    v8bf b1lo = *(const v8bf*)bp1, b1hi = *(const v8bf*)(bp1 + 16);
    v16bf bb0, bb1;
    #pragma unroll
    for (int i = 0; i < 8; ++i) {
      bb0[i] = b0lo[i]; bb0[8 + i] = b0hi[i];
      bb1[i] = b1lo[i]; bb1[8 + i] = b1hi[i];
    }
    acc0 = __builtin_amdgcn_wmma_f32_16x16x32_bf16(false, a, false, bb0,
                                                   (short)0, acc0, false, false);
    acc1 = __builtin_amdgcn_wmma_f32_16x16x32_bf16(false, a, false, bb1,
                                                   (short)0, acc1, false, false);
  }

  // Epilogue: lanes 0-15 -> M = mbase+i, lanes 16-31 -> M = mbase+8+i
  int nl    = lane & 15;
  int mhalf = (lane >> 4) * 8;
  if constexpr (BF16OUT) {
    bf16* yb = (bf16*)Yv + (long)b * CCH * HWPIX;
    #pragma unroll
    for (int i = 0; i < 8; ++i) {
      int m = mbase + mhalf + i;
      long o0 = (long)m * HWPIX + n0 + nl;
      yb[o0]      = (bf16)acc0[i];
      yb[o0 + 16] = (bf16)acc1[i];
    }
  } else {
    float* yb = (float*)Yv + (long)b * CCH * HWPIX;
    const float* rb = resid + (long)b * CCH * HWPIX;
    #pragma unroll
    for (int i = 0; i < 8; ++i) {
      int m = mbase + mhalf + i;
      long o0 = (long)m * HWPIX + n0 + nl;
      yb[o0]      = acc0[i] + rb[o0];
      yb[o0 + 16] = acc1[i] + rb[o0 + 16];
    }
  }
}

// ---------------------------------------------------------------------------
// Depthwise 3x3 SAME conv (bf16 in -> bf16 out) with optional fused sum(x^2)
// per (b,c) for the downstream l2 normalization (block reduce + one atomic).
// ---------------------------------------------------------------------------
__global__ void __launch_bounds__(256)
dwconv3x3(const bf16* __restrict__ Xin, const float* __restrict__ Wdw,
          bf16* __restrict__ Yout, float* __restrict__ sumsq) {
  long g = (long)blockIdx.x * 256 + threadIdx.x;    // B*C*HW threads
  int w = (int)(g % WW);
  long t = g / WW;
  int h = (int)(t % HH); t /= HH;
  int c = (int)(t % CCH);
  int b = (int)(t / CCH);

  const bf16*  xp = Xin + ((long)b * CCH + c) * HWPIX;
  const float* wp = Wdw + c * 9;
  float acc = 0.f;
  #pragma unroll
  for (int dy = -1; dy <= 1; ++dy) {
    int hy = h + dy;
    if (hy < 0 || hy >= HH) continue;
    #pragma unroll
    for (int dx = -1; dx <= 1; ++dx) {
      int wx = w + dx;
      if (wx < 0 || wx >= WW) continue;
      acc += wp[(dy + 1) * 3 + (dx + 1)] * (float)xp[hy * WW + wx];
    }
  }
  Yout[g] = (bf16)acc;

  if (sumsq) {       // whole block shares one (b,c): HW % 256 == 0
    float v = acc * acc;
    #pragma unroll
    for (int off = 16; off; off >>= 1) v += __shfl_down(v, off, 32);
    __shared__ float red[8];
    int lane = threadIdx.x & 31, wv = threadIdx.x >> 5;
    if (lane == 0) red[wv] = v;
    __syncthreads();
    if (threadIdx.x == 0) {
      float s = 0.f;
      #pragma unroll
      for (int i = 0; i < 8; ++i) s += red[i];
      atomicAdd(&sumsq[b * CCH + c], s);
    }
  }
}

// ---------------------------------------------------------------------------
// Split-K channel-attention logits: partial (q_c . k_e) over 256-pixel chunk,
// atomically accumulated into zero-initialized logits. grid = (HW/256, B*H).
// ---------------------------------------------------------------------------
__global__ void __launch_bounds__(256)
attn_partial(const bf16* __restrict__ Q, const bf16* __restrict__ K,
             float* __restrict__ logits) {
  int bh = blockIdx.y;
  int b = bh / HEADS, h = bh % HEADS;
  int p0 = blockIdx.x * 256;
  const bf16* qb = Q + ((long)b * CCH + h * DC) * HWPIX + p0;
  const bf16* kb = K + ((long)b * CCH + h * DC) * HWPIX + p0;

  __shared__ bf16 qs[DC * 256];   // 16 KB
  __shared__ bf16 ks[DC * 256];   // 16 KB
  int tid = threadIdx.x;
  for (int i = tid; i < DC * 256; i += 256) {
    int cc = i >> 8, pp = i & 255;
    qs[i] = qb[(long)cc * HWPIX + pp];
    ks[i] = kb[(long)cc * HWPIX + pp];
  }
  __syncthreads();

  #pragma unroll
  for (int it = 0; it < 4; ++it) {
    int pi = it * 256 + tid;
    int c = pi >> 5, e = pi & 31;
    const bf16* qr = &qs[c * 256];
    const bf16* kr = &ks[e * 256];
    float a = 0.f;
    for (int pp = 0; pp < 256; pp += 8) {
      v8bf q8 = *(const v8bf*)(qr + pp);
      v8bf k8 = *(const v8bf*)(kr + pp);
      #pragma unroll
      for (int j = 0; j < 8; ++j) a += (float)q8[j] * (float)k8[j];
    }
    atomicAdd(&logits[((long)bh * DC + c) * DC + e], a);
  }
}

// Row softmax over last dim (32) with fused l2-norm + temperature scaling.
__global__ void softmax32(float* __restrict__ logits,
                          const float* __restrict__ ssq,
                          const float* __restrict__ ssk,
                          const float* __restrict__ temp) {
  int r = blockIdx.x * 256 + threadIdx.x;
  if (r >= BATCH * HEADS * DC) return;
  int bh = r / DC, c = r % DC;
  int b = bh / HEADS, h = bh % HEADS;
  float nq = fmaxf(sqrtf(ssq[b * CCH + h * DC + c]), EPS_NORM);
  float tpr = temp[h] / nq;

  float* row = logits + (long)r * DC;
  float vals[DC];
  float m = -1e30f;
  for (int e = 0; e < DC; ++e) {
    float nk = fmaxf(sqrtf(ssk[b * CCH + h * DC + e]), EPS_NORM);
    vals[e] = row[e] * tpr / nk;
    m = fmaxf(m, vals[e]);
  }
  float s = 0.f;
  for (int e = 0; e < DC; ++e) { vals[e] = __expf(vals[e] - m); s += vals[e]; }
  float inv = 1.f / s;
  for (int e = 0; e < DC; ++e) row[e] = vals[e] * inv;
}

// out[b, h*32+c, p] = sum_e attn[b,h,c,e] * v[b, h*32+e, p]  (bf16 out)
__global__ void __launch_bounds__(256)
attn_apply(const float* __restrict__ attn, const bf16* __restrict__ V,
           bf16* __restrict__ O) {
  int bh = blockIdx.y;
  int b = bh / HEADS, h = bh % HEADS;
  long p = (long)blockIdx.x * 256 + threadIdx.x;

  __shared__ float at[DC * DC];
  for (int i = threadIdx.x; i < DC * DC; i += 256)
    at[i] = attn[(long)bh * DC * DC + i];
  __syncthreads();

  const bf16* vb = V + ((long)b * CCH + h * DC) * HWPIX + p;
  bf16*       ob = O + ((long)b * CCH + h * DC) * HWPIX + p;
  float vv[DC];
  #pragma unroll
  for (int e = 0; e < DC; ++e) vv[e] = (float)vb[(long)e * HWPIX];
  #pragma unroll 4
  for (int c = 0; c < DC; ++c) {
    const float* ar = &at[c * DC];
    float a = 0.f;
    #pragma unroll
    for (int e = 0; e < DC; ++e) a += ar[e] * vv[e];
    ob[(long)c * HWPIX] = (bf16)a;
  }
}

// ---------------------------------------------------------------------------
extern "C" void kernel_launch(void* const* d_in, const int* in_sizes, int n_in,
                              void* d_out, int out_size, void* d_ws, size_t ws_size,
                              hipStream_t stream) {
  const float* state = (const float*)d_in[0];
  const float* dark  = (const float*)d_in[1];
  const float* nq_w  = (const float*)d_in[2];
  const float* nq_b  = (const float*)d_in[3];
  const float* nkv_w = (const float*)d_in[4];
  const float* nkv_b = (const float*)d_in[5];
  const float* q_pw  = (const float*)d_in[6];
  const float* q_dw  = (const float*)d_in[7];
  const float* k_pw  = (const float*)d_in[8];
  const float* k_dw  = (const float*)d_in[9];
  const float* v_pw  = (const float*)d_in[10];
  const float* v_dw  = (const float*)d_in[11];
  const float* proj_w = (const float*)d_in[12];
  const float* temperature = (const float*)d_in[13];
  float* out = (float*)d_out;

  uint8_t* ws = (uint8_t*)d_ws;
  size_t off = 0;
  auto carve = [&](size_t bytes) -> void* {
    void* p = (void*)(ws + off);
    off += (bytes + 255) & ~(size_t)255;
    return p;
  };

  const size_t NELEM = (size_t)BATCH * CCH * HWPIX;  // 16.7M
  bf16*  wpack  = (bf16*)carve(4 * 128 * 128 * sizeof(bf16));
  float* norms  = (float*)carve(2 * BATCH * CCH * sizeof(float)); // [q | k] sumsq
  float* logits = (float*)carve((size_t)BATCH * HEADS * DC * DC * sizeof(float));
  bf16*  XQ     = (bf16*)carve(NELEM * sizeof(bf16));
  bf16*  XK     = (bf16*)carve(NELEM * sizeof(bf16));
  bf16*  XV     = (bf16*)carve(NELEM * sizeof(bf16));
  bf16*  G      = (bf16*)carve(NELEM * sizeof(bf16));

  // 1. pack weights to bf16, zero accumulators
  pack_w<<<64, 256, 0, stream>>>(q_pw, k_pw, v_pw, proj_w, wpack);
  zero_f32<<<(2 * BATCH * CCH + 255) / 256, 256, 0, stream>>>(norms, 2 * BATCH * CCH);
  zero_f32<<<(BATCH * HEADS * DC * DC + 255) / 256, 256, 0, stream>>>(
      logits, BATCH * HEADS * DC * DC);

  // 2. LayerNorm + bf16 packing (single global read via LDS tiles)
  ln_pack<<<BATCH * HWPIX / TPX, 256, 0, stream>>>(
      state, dark, nq_w, nq_b, nkv_w, nkv_b, XQ, XK, XV);

  dim3 gg(HWPIX / 32, BATCH);
  long dwgrid = (long)NELEM / 256;

  // 3. q = dw(conv1x1(LN(state)))  — fused sumsq for l2 norm
  gemm_c128<true><<<gg, 256, 0, stream>>>(wpack + 0 * 16384, XQ, (const float*)0, G);
  dwconv3x3<<<dwgrid, 256, 0, stream>>>(G, q_dw, XQ, norms);

  // 4. k = dw(conv1x1(LN(dark)))
  gemm_c128<true><<<gg, 256, 0, stream>>>(wpack + 1 * 16384, XK, (const float*)0, G);
  dwconv3x3<<<dwgrid, 256, 0, stream>>>(G, k_dw, XK, norms + BATCH * CCH);

  // 5. v = dw(conv1x1(dark))
  gemm_c128<true><<<gg, 256, 0, stream>>>(wpack + 2 * 16384, XV, (const float*)0, G);
  dwconv3x3<<<dwgrid, 256, 0, stream>>>(G, v_dw, XV, (float*)0);

  // 6. channel attention: split-K logits -> scaled softmax -> attn @ v
  attn_partial<<<dim3(HWPIX / 256, BATCH * HEADS), 256, 0, stream>>>(XQ, XK, logits);
  softmax32<<<(BATCH * HEADS * DC + 255) / 256, 256, 0, stream>>>(
      logits, norms, norms + BATCH * CCH, temperature);
  attn_apply<<<dim3(HWPIX / 256, BATCH * HEADS), 256, 0, stream>>>(logits, XV, XQ);

  // 7. proj conv1x1 + residual -> f32 output
  gemm_c128<false><<<gg, 256, 0, stream>>>(wpack + 3 * 16384, XQ, state, out);
}
